// ScaledDotProductAttention_33543694582533
// MI455X (gfx1250) — matrix operand order repaired and checked by reading, not compiled, not run on hardware
//
#include <hip/hip_runtime.h>

// ---------- CDNA5 WMMA types/helpers ----------
typedef __attribute__((ext_vector_type(16))) __bf16 v16bf;
typedef __attribute__((ext_vector_type(8)))  float  v8f;

union F16 { v16bf v; unsigned int u[8]; };
union U8x16 { unsigned short h[8]; uint4 q; };

__device__ __forceinline__ unsigned short f2bf(float f) {
  unsigned int u = __float_as_uint(f);
  u += 0x7FFFu + ((u >> 16) & 1u);       // round-to-nearest-even
  return (unsigned short)(u >> 16);
}
__device__ __forceinline__ float bf2f(unsigned short h) {
  return __uint_as_float(((unsigned int)h) << 16);
}
__device__ __forceinline__ v8f wmma_bf16(v16bf a, v16bf b, v8f c) {
  return __builtin_amdgcn_wmma_f32_16x16x32_bf16(false, a, false, b, (short)0, c,
                                                 false, false);
}
__device__ __forceinline__ v8f v8f_zero() {
  v8f z;
#pragma unroll
  for (int i = 0; i < 8; ++i) z[i] = 0.0f;
  return z;
}

// ---------- CDNA5 async global->LDS copy (ASYNCcnt path) ----------
__device__ __forceinline__ unsigned lds_off_of(const void* p) {
  return (unsigned)(unsigned long long)(__attribute__((address_space(3))) const char*)p;
}
__device__ __forceinline__ void async_copy16(unsigned lds_off, const void* gaddr) {
  asm volatile("global_load_async_to_lds_b128 %0, %1, off"
               :
               : "v"(lds_off), "v"((unsigned long long)(size_t)gaddr)
               : "memory");
}
__device__ __forceinline__ void wait_async0() {
  asm volatile("s_wait_asynccnt 0" ::: "memory");
}

#define NB 2
#define LSEQ 1024
#define NH 16
#define HD 64
#define DIMF 1024
#define RPE_P 257   // 2*128+1

// ---------- fp32 -> bf16 convert (rpe table) ----------
__global__ void cvt_bf16_kernel(const float* __restrict__ src,
                                unsigned short* __restrict__ dst, int n) {
  int i = blockIdx.x * blockDim.x + threadIdx.x;
  if (i < n) dst[i] = f2bf(src[i]);
}

// ---------- projection GEMM: Out = X (Mx1024) @ W^T (1024x1024), bf16 head layout ----------
// block = 256 threads = 8 waves (4x2), block tile 64(M) x 128(N), K step 32
__global__ __launch_bounds__(256) void proj_kernel(
    const float* __restrict__ X,            // [2048][1024]
    const float* __restrict__ W,            // [1024][1024] (row n holds W[n][k])
    unsigned short* __restrict__ Out)       // [NB*NH][LSEQ][HD] bf16
{
  __shared__ __align__(16) unsigned short atile[64 * 32];
  __shared__ __align__(16) unsigned short wtile[128 * 32];
  const int tid = threadIdx.x;
  const int wid = tid >> 5, lane = tid & 31;
  const int h2 = lane >> 4, nl = lane & 15;
  const int wm = wid & 3, wn = wid >> 2;
  const int bm = blockIdx.y * 64;
  const int bn = blockIdx.x * 128;

  v8f acc[4];
#pragma unroll
  for (int t = 0; t < 4; ++t) acc[t] = v8f_zero();

  const int srow = tid >> 2;          // 0..63
  const int scol = (tid & 3) * 8;     // 0,8,16,24

  for (int k0 = 0; k0 < DIMF; k0 += 32) {
    { // stage A tile (64x32) fp32->bf16
      const float* src = X + (size_t)(bm + srow) * DIMF + k0 + scol;
      U8x16 t;
#pragma unroll
      for (int j = 0; j < 8; ++j) t.h[j] = f2bf(src[j]);
      *(uint4*)(atile + srow * 32 + scol) = t.q;
    }
#pragma unroll
    for (int r = 0; r < 2; ++r) { // stage W tile (128x32)
      const int row = srow + 64 * r;
      const float* src = W + (size_t)(bn + row) * DIMF + k0 + scol;
      U8x16 t;
#pragma unroll
      for (int j = 0; j < 8; ++j) t.h[j] = f2bf(src[j]);
      *(uint4*)(wtile + row * 32 + scol) = t.q;
    }
    __syncthreads();

    F16 af; // A fragment: rows 16*wm..+15, K = k0..k0+31
#pragma unroll
    for (int v = 0; v < 8; ++v) {
      const int kk = 16 * h2 + 2 * v;
      af.u[v] = *(const unsigned int*)(atile + (16 * wm + nl) * 32 + kk);
    }
#pragma unroll
    for (int nt = 0; nt < 4; ++nt) { // B fragments: B[k][n] = W[n][k]
      F16 bfm;
#pragma unroll
      for (int v = 0; v < 8; ++v)
        bfm.u[v] = *(const unsigned int*)(wtile + (64 * wn + 16 * nt + nl) * 32 +
                                          16 * h2 + 2 * v);
      acc[nt] = wmma_bf16(af.v, bfm.v, acc[nt]);
    }
    __syncthreads();
  }

  // store bf16 in [batch*NH + h][l][dd] layout
#pragma unroll
  for (int v = 0; v < 8; ++v) {
    const int mg = bm + 16 * wm + v + 8 * h2;
    const int batch = mg >> 10, l = mg & 1023;
#pragma unroll
    for (int nt = 0; nt < 4; ++nt) {
      const int ng = bn + 64 * wn + 16 * nt + nl;
      const int h = ng >> 6, dd = ng & 63;
      Out[(((size_t)(batch * NH + h) * LSEQ) + l) * HD + dd] = f2bf(acc[nt][v]);
    }
  }
}

// ---------- RPE table: T[bh][p][k] = (P[p,h,:] . K[bh,k,:]) / 8, bf16 ----------
// one wave per 16x16 tile; contraction pairs contiguous in both operands -> direct global loads
__global__ __launch_bounds__(128) void rpet_kernel(
    const unsigned short* __restrict__ Pb,  // [257][1024] bf16
    const unsigned short* __restrict__ Kh,  // [NB*NH][LSEQ][HD] bf16
    unsigned short* __restrict__ Tb)        // [NB*NH][257][LSEQ] bf16
{
  const int tid = threadIdx.x, wid = tid >> 5, lane = tid & 31;
  const int h2 = lane >> 4, nl = lane & 15;
  const int bh = blockIdx.x;
  const int head = bh & (NH - 1);
  const int id = blockIdx.y * 4 + wid;      // 0..1087
  if (id >= 17 * 64) return;
  const int p0 = (id >> 6) * 16, kk0 = (id & 63) * 16;
  const unsigned short* Kb = Kh + (size_t)bh * (LSEQ * HD);

  v8f acc = v8f_zero();
  int prow = p0 + nl; if (prow > 256) prow = 256;
#pragma unroll
  for (int g = 0; g < 2; ++g) {
    F16 af, bfm;
#pragma unroll
    for (int v = 0; v < 8; ++v) {
      const int kk = 16 * h2 + 2 * v;
      af.u[v] = *(const unsigned int*)(Pb + (size_t)prow * DIMF + head * HD +
                                       32 * g + kk);
    }
#pragma unroll
    for (int v = 0; v < 8; ++v)
      bfm.u[v] = *(const unsigned int*)(Kb + (size_t)(kk0 + nl) * HD + 32 * g +
                                        16 * h2 + 2 * v);
    acc = wmma_bf16(af.v, bfm.v, acc);
  }
  unsigned short* Trow = Tb + (size_t)bh * (RPE_P * LSEQ);
#pragma unroll
  for (int v = 0; v < 8; ++v) {
    const int p = p0 + v + 8 * h2;
    if (p < RPE_P) Trow[(size_t)p * LSEQ + kk0 + nl] = f2bf(acc[v] * 0.125f);
  }
}

// ---------- flash attention with fused RPE bias + causal + key mask ----------
// block = 128 threads = 4 waves; one wave per 16-query tile; key step 32
// K/V tiles staged via CDNA5 async global->LDS copies (ASYNCcnt), no VGPR bounce
__global__ __launch_bounds__(128) void attn_kernel(
    const unsigned short* __restrict__ Qh, const unsigned short* __restrict__ Kh,
    const unsigned short* __restrict__ Vh, const unsigned short* __restrict__ Tb,
    const unsigned char* __restrict__ qmask, const unsigned char* __restrict__ kmask,
    float* __restrict__ out)
{
  __shared__ __align__(16) unsigned short smem[4 * 4608]; // per wave: K 32x64, V 32x64, P 16x32
  const int tid = threadIdx.x;
  const int wid = tid >> 5, lane = tid & 31;
  const int h2 = lane >> 4, nl = lane & 15;
  const int bh = blockIdx.x;                 // batch*NH + head
  const int batch = bh >> 4, head = bh & 15;
  const int q0 = (blockIdx.y * 4 + wid) * 16;

  const unsigned short* Qb = Qh + (size_t)bh * (LSEQ * HD);
  const unsigned short* Kb = Kh + (size_t)bh * (LSEQ * HD);
  const unsigned short* Vb = Vh + (size_t)bh * (LSEQ * HD);
  const unsigned short* Tp = Tb + (size_t)bh * (RPE_P * LSEQ);
  const unsigned char* km = kmask + batch * LSEQ;

  unsigned short* kb = smem + wid * 4608;
  unsigned short* vb = kb + 2048;
  unsigned short* pb = vb + 2048;
  const unsigned kboff = lds_off_of(kb + lane * HD);  // this lane's K row in LDS
  const unsigned vboff = lds_off_of(vb + lane * HD);  // this lane's V row in LDS

  // Q as two A fragments (dd chunks of 32), loaded once
  F16 aq[2];
#pragma unroll
  for (int g = 0; g < 2; ++g)
#pragma unroll
    for (int v = 0; v < 8; ++v) {
      const int kk = 16 * h2 + 2 * v;
      aq[g].u[v] = *(const unsigned int*)(Qb + (size_t)(q0 + nl) * HD + 32 * g + kk);
    }

  v8f accO[4];
  float mrow[8], lrow[8];
#pragma unroll
  for (int t = 0; t < 4; ++t) accO[t] = v8f_zero();
#pragma unroll
  for (int v = 0; v < 8; ++v) { mrow[v] = -1e30f; lrow[v] = 0.0f; }

  for (int k0 = 0; k0 <= q0 + 15; k0 += 32) {
    // stage K/V tiles (32 rows x 64 bf16) straight into LDS, one row per lane
    {
      const unsigned short* kg = Kb + (size_t)(k0 + lane) * HD;
      const unsigned short* vg = Vb + (size_t)(k0 + lane) * HD;
#pragma unroll
      for (int j = 0; j < 8; ++j) {
        async_copy16(kboff + 16 * j, kg + 8 * j);
        async_copy16(vboff + 16 * j, vg + 8 * j);
      }
      wait_async0();
    }

    // S = Q K^T for two 16-key subtiles (B[k=dd][n=key] read transposed from LDS)
    v8f S[2];
#pragma unroll
    for (int c = 0; c < 2; ++c) {
      F16 b0, b1;
      const unsigned short* krow = kb + (16 * c + nl) * HD;
#pragma unroll
      for (int v = 0; v < 8; ++v) {
        b0.u[v] = *(const unsigned int*)(krow + 16 * h2 + 2 * v);
        b1.u[v] = *(const unsigned int*)(krow + 32 + 16 * h2 + 2 * v);
      }
      S[c] = wmma_bf16(aq[0].v, b0.v, v8f_zero());
      S[c] = wmma_bf16(aq[1].v, b1.v, S[c]);
    }

    // fused RPE bias gather + causal + key-mask, then online softmax stats
    const int kA = k0 + nl, kBk = k0 + 16 + nl;
    const int km0 = km[kA] ? 1 : 0, km1 = km[kBk] ? 1 : 0;
    float pr[2][8], newmax[8];
#pragma unroll
    for (int v = 0; v < 8; ++v) {
      const int q = q0 + v + 8 * h2;
      float s0 = S[0][v] * 0.125f, s1 = S[1][v] * 0.125f;
      int i0 = 128 + kA - q;  i0 = i0 < 0 ? 0 : (i0 > 256 ? 256 : i0);
      int i1 = 128 + kBk - q; i1 = i1 < 0 ? 0 : (i1 > 256 ? 256 : i1);
      s0 += bf2f(Tp[(size_t)i0 * LSEQ + kA]);
      s1 += bf2f(Tp[(size_t)i1 * LSEQ + kBk]);
      if (kA > q || km0) s0 = -1e30f;
      if (kBk > q || km1) s1 = -1e30f;
      pr[0][v] = s0; pr[1][v] = s1;
      float t = fmaxf(s0, s1);
      t = fmaxf(t, __shfl_xor(t, 1));
      t = fmaxf(t, __shfl_xor(t, 2));
      t = fmaxf(t, __shfl_xor(t, 4));
      t = fmaxf(t, __shfl_xor(t, 8));
      newmax[v] = t;
    }
#pragma unroll
    for (int v = 0; v < 8; ++v) {
      const float mn = fmaxf(mrow[v], newmax[v]);
      const float corr = __expf(mrow[v] - mn);
      mrow[v] = mn;
      const float p0 = __expf(pr[0][v] - mn);
      const float p1 = __expf(pr[1][v] - mn);
      float rs = p0 + p1;
      rs += __shfl_xor(rs, 1); rs += __shfl_xor(rs, 2);
      rs += __shfl_xor(rs, 4); rs += __shfl_xor(rs, 8);
      lrow[v] = lrow[v] * corr + rs;
#pragma unroll
      for (int t = 0; t < 4; ++t) accO[t][v] *= corr;
      pb[(v + 8 * h2) * 32 + nl] = f2bf(p0);       // re-layout P via LDS
      pb[(v + 8 * h2) * 32 + 16 + nl] = f2bf(p1);
    }

    // read P back as 16x32 A fragment
    F16 pf;
#pragma unroll
    for (int v = 0; v < 8; ++v) {
      const int kk = 16 * h2 + 2 * v;
      pf.u[v] = *(const unsigned int*)(pb + nl * 32 + kk);
    }
    // O += P @ V  (B[k=key][n=dd] straight from LDS rows)
#pragma unroll
    for (int t = 0; t < 4; ++t) {
      F16 bv;
#pragma unroll
      for (int v = 0; v < 8; ++v) {
        const int kkl = 16 * h2 + 2 * v;
        const unsigned int e0 = vb[kkl * HD + 16 * t + nl];
        const unsigned int e1 = vb[(kkl + 1) * HD + 16 * t + nl];
        bv.u[v] = e0 | (e1 << 16);
      }
      accO[t] = wmma_bf16(pf.v, bv.v, accO[t]);
    }
  }

  // epilogue: normalize, apply query mask, store fp32
#pragma unroll
  for (int v = 0; v < 8; ++v) {
    const int q = q0 + v + 8 * h2;
    const float inv = 1.0f / lrow[v];
    const int qz = qmask[batch * LSEQ + q] ? 1 : 0;
    float* orow = out + ((size_t)batch * LSEQ + q) * DIMF + head * HD;
#pragma unroll
    for (int t = 0; t < 4; ++t)
      orow[16 * t + nl] = qz ? 0.0f : accO[t][v] * inv;
  }
}

// ---------- launch ----------
extern "C" void kernel_launch(void* const* d_in, const int* in_sizes, int n_in,
                              void* d_out, int out_size, void* d_ws, size_t ws_size,
                              hipStream_t stream) {
  (void)in_sizes; (void)n_in; (void)out_size; (void)ws_size;
  const float* query = (const float*)d_in[0];
  const float* key   = (const float*)d_in[1];
  const unsigned char* qmask = (const unsigned char*)d_in[2];
  const unsigned char* kmask = (const unsigned char*)d_in[3];
  const float* Wq  = (const float*)d_in[4];
  const float* Wk  = (const float*)d_in[5];
  const float* Wv  = (const float*)d_in[6];
  const float* rpe = (const float*)d_in[7];
  float* out = (float*)d_out;

  char* ws = (char*)d_ws;
  const size_t projBytes = (size_t)NB * NH * LSEQ * HD * 2; // 4 MiB each
  unsigned short* Qh = (unsigned short*)ws; ws += projBytes;
  unsigned short* Kh = (unsigned short*)ws; ws += projBytes;
  unsigned short* Vh = (unsigned short*)ws; ws += projBytes;
  unsigned short* Pb = (unsigned short*)ws; ws += (size_t)RPE_P * DIMF * 2;
  unsigned short* Tb = (unsigned short*)ws; // NB*NH*257*1024*2 ~ 17 MB

  const int nCvt = RPE_P * DIMF;
  cvt_bf16_kernel<<<(nCvt + 255) / 256, 256, 0, stream>>>(rpe, Pb, nCvt);

  dim3 pg(DIMF / 128, (NB * LSEQ) / 64); // (8, 32)
  proj_kernel<<<pg, 256, 0, stream>>>(query, Wq, Qh);
  proj_kernel<<<pg, 256, 0, stream>>>(key,   Wk, Kh);
  proj_kernel<<<pg, 256, 0, stream>>>(key,   Wv, Vh);

  rpet_kernel<<<dim3(NB * NH, 272), 128, 0, stream>>>(Pb, Kh, Tb);

  attn_kernel<<<dim3(NB * NH, LSEQ / 64), 128, 0, stream>>>(Qh, Kh, Vh, Tb,
                                                            qmask, kmask, out);
}
